// KPConvSimple_68281390072560
// MI455X (gfx1250) — matrix operand order repaired and checked by reading, not compiled, not run on hardware
//
#include <hip/hip_runtime.h>

#define B_   8
#define N_   4096
#define KNN  16
#define CIN  64
#define COUT 128
#define MROWS (B_*N_*KNN)   // 524288 rows for the edge GEMMs
#define EDGE_K (2*CIN+3)    // 131
#define EDGE_KP 160         // padded to 5 * 32

typedef __attribute__((ext_vector_type(16))) _Float16 v16h;
typedef __attribute__((ext_vector_type(8)))  _Float16 v8h;
typedef __attribute__((ext_vector_type(8)))  float    v8f;
typedef __attribute__((ext_vector_type(4)))  unsigned int u32x4;
typedef __attribute__((ext_vector_type(4)))  int          i32x4;
typedef __attribute__((ext_vector_type(8)))  int          i32x8;

#if defined(__has_builtin)
#if __has_builtin(__builtin_amdgcn_tensor_load_to_lds) && \
    __has_builtin(__builtin_amdgcn_s_wait_tensorcnt)
#define USE_TDM 1
#endif
#endif
#ifndef USE_TDM
#define USE_TDM 0
#endif

// ---------------------------------------------------------------------------
// Stage `bytes` of contiguous weights from global into LDS.
// Preferred path: Tensor Data Mover (TENSOR_LOAD_TO_LDS) with a 1-D D#:
//   group0: count=1 | lds_addr | global_addr[56:0] | type=2
//   group1: data_size=8B, tensor_dim0 = tile_dim0 = stride = bytes/8,
//           tensor_dim1 = tile_dim1 = 1, workgroup_mask = 0 (not in cluster)
// Wave 0 issues the DMA (EXEC ignored, wave-level op) and waits TENSORcnt==0;
// caller's __syncthreads() publishes the LDS tile to all waves.
// ---------------------------------------------------------------------------
__device__ inline void stage_weights(const _Float16* __restrict__ wsw,
                                     _Float16* sB, int bytes, int tid) {
#if USE_TDM
  if (tid < 32) {
    const unsigned elems = (unsigned)bytes >> 3;  // 8-byte units
    const unsigned long long ga = (unsigned long long)(uintptr_t)wsw;
    const unsigned lds = (unsigned)(uintptr_t)sB; // LDS offset = low 32 bits
    u32x4 g0;
    g0[0] = 1u;                                       // count=1, user mode
    g0[1] = lds;                                      // lds_addr
    g0[2] = (unsigned)(ga & 0xFFFFFFFFu);             // global_addr[31:0]
    g0[3] = (unsigned)((ga >> 32) & 0x01FFFFFFu) | (2u << 30);  // [56:32]|type=2
    i32x8 g1;
    g1[0] = 0x00030000;           // workgroup_mask=0, data_size=3 (8 bytes)
    g1[1] = (int)(elems << 16);   // tensor_dim0[15:0] (elems < 2^16)
    g1[2] = (int)(1u << 16);      // tensor_dim0 hi = 0, tensor_dim1 = 1
    g1[3] = (int)(elems << 16);   // tensor_dim1 hi = 0, tile_dim0 = elems
    g1[4] = 1;                    // tile_dim1 = 1, tile_dim2 = 0
    g1[5] = (int)elems;           // tensor_dim0_stride lo
    g1[6] = 0;                    // stride hi, tensor_dim1_stride lo
    g1[7] = 0;
    const i32x4 z4 = {0, 0, 0, 0};
#if __clang_major__ >= 23
    const i32x8 z8 = {0, 0, 0, 0, 0, 0, 0, 0};
    __builtin_amdgcn_tensor_load_to_lds(g0, g1, z4, z4, z8, 0);
#else
    __builtin_amdgcn_tensor_load_to_lds(g0, g1, z4, z4, 0);
#endif
    __builtin_amdgcn_s_wait_tensorcnt((short)0);
  }
#else
  const int n8 = bytes >> 4;  // 16-byte chunks
  for (int i = tid; i < n8; i += 256)
    ((v8h*)sB)[i] = ((const v8h*)wsw)[i];
#endif
}

// ---------------------------------------------------------------------------
// WMMA tile engine. sA is row-major [128 x LDA] f16. sBsw is pre-swizzled into
// per-lane fragment order: element ((kt*NCT+ct)*32 + lane)*16 + e  holds
// B[k = kt*32 + (lane>>4)*16 + e][n = ct*16 + (lane&15)], so a B fragment is
// one contiguous 32-byte v16h load (2x ds_load_b128). The A fragment is two
// contiguous 16-byte runs (k = half*16 + e, matching B) -> two v8h loads + shuffle.
// ---------------------------------------------------------------------------
template <int NCT, int KSTEPS, int LDA>
__device__ inline void wmma_block(const _Float16* sA, const _Float16* sBsw,
                                  int wv, int lane, v8f acc[NCT]) {
  const int mrow = lane & 15;
  const int half = lane >> 4;
#pragma unroll
  for (int kt = 0; kt < KSTEPS; ++kt) {
    const _Float16* ap = sA + (wv * 16 + mrow) * LDA + kt * 32 + half * 16;
    const v8h alo = *(const v8h*)ap;
    const v8h ahi = *(const v8h*)(ap + 8);
    const v16h afrag = __builtin_shufflevector(
        alo, ahi, 0, 1, 2, 3, 4, 5, 6, 7, 8, 9, 10, 11, 12, 13, 14, 15);
#pragma unroll
    for (int ct = 0; ct < NCT; ++ct) {
      const v16h bfrag =
          *(const v16h*)(sBsw + (((kt * NCT + ct) * 32) + lane) * 16);
      acc[ct] = __builtin_amdgcn_wmma_f32_16x16x32_f16(
          false, afrag, false, bfrag, (short)0, acc[ct], false, false);
    }
  }
}

// ---------------------------------------------------------------------------
// zero the BN statistics scratch
// ---------------------------------------------------------------------------
__global__ void k_zero(float* p, int n) {
  int i = blockIdx.x * blockDim.x + threadIdx.x;
  if (i < n) p[i] = 0.f;
}

// ---------------------------------------------------------------------------
// Pre-swizzle a weight matrix W[COUTxKreal] (f32, row-major) into the
// per-lane f16 fragment layout described above. Runs once.
// ---------------------------------------------------------------------------
__global__ void k_prepw(const float* __restrict__ w, _Float16* __restrict__ wsw,
                        int NCT, int Kreal, int total) {
  int t = blockIdx.x * blockDim.x + threadIdx.x;
  if (t >= total) return;
  const int e    = t & 15;
  const int lane = (t >> 4) & 31;
  const int ctt  = t >> 9;
  const int ct   = ctt % NCT;
  const int kt   = ctt / NCT;
  const int k    = kt * 32 + (lane >> 4) * 16 + e;
  const int n    = ct * 16 + (lane & 15);
  wsw[t] = (k < Kreal) ? (_Float16)w[(size_t)n * Kreal + k] : (_Float16)0.f;
}

// ---------------------------------------------------------------------------
// features [B,CIN,N] (f32) -> featT [B,N,CIN] (f16)
// ---------------------------------------------------------------------------
__global__ void k_transpose(const float* __restrict__ feat,
                            _Float16* __restrict__ featT) {
  int e = blockIdx.x * blockDim.x + threadIdx.x;
  if (e >= B_ * N_ * CIN) return;
  int b = e / (N_ * CIN);
  int r = e % (N_ * CIN);
  int n = r / CIN;
  int c = r % CIN;
  featT[e] = (_Float16)feat[((size_t)b * CIN + c) * N_ + n];
}

// ---------------------------------------------------------------------------
// Brute-force KNN: register top-16 with max tracking (fully unrolled).
// ---------------------------------------------------------------------------
__global__ __launch_bounds__(256) void k_knn(const float* __restrict__ xyz,
                                             int* __restrict__ idxOut) {
  __shared__ float sx[256], sy[256], sz[256];
  const int tid = threadIdx.x;
  const int b   = blockIdx.x >> 4;                 // N_/256 == 16 tiles/batch
  const int n   = ((blockIdx.x & 15) << 8) + tid;
  const float* xb = xyz + (size_t)b * 3 * N_;
  const float px = xb[n], py = xb[N_ + n], pz = xb[2 * N_ + n];

  float d[KNN]; int id[KNN];
#pragma unroll
  for (int i = 0; i < KNN; ++i) { d[i] = 3.0e38f; id[i] = 0; }
  float maxv = 3.0e38f; int maxpos = 0;

  for (int mt = 0; mt < N_ / 256; ++mt) {
    const int m = mt * 256 + tid;
    sx[tid] = xb[m]; sy[tid] = xb[N_ + m]; sz[tid] = xb[2 * N_ + m];
    __syncthreads();
    for (int j = 0; j < 256; ++j) {
      const float dx = sx[j] - px, dy = sy[j] - py, dz = sz[j] - pz;
      const float c = dx * dx + dy * dy + dz * dz;
      if (c < maxv) {
#pragma unroll
        for (int i = 0; i < KNN; ++i)
          if (i == maxpos) { d[i] = c; id[i] = mt * 256 + j; }
        maxv = d[0]; maxpos = 0;
#pragma unroll
        for (int i = 1; i < KNN; ++i)
          if (d[i] > maxv) { maxv = d[i]; maxpos = i; }
      }
    }
    __syncthreads();
  }
  int* o = idxOut + ((size_t)b * N_ + n) * KNN;
#pragma unroll
  for (int i = 0; i < KNN; ++i) o[i] = id[i];
}

// ---------------------------------------------------------------------------
// GEMM1: edge[M x 131(pad 160)] @ w1^T -> y1 (f16) + per-channel sum/sumsq.
// ---------------------------------------------------------------------------
__global__ __launch_bounds__(256) void k_gemm1(
    const _Float16* __restrict__ featT, const float* __restrict__ xyz,
    const int* __restrict__ idx, const _Float16* __restrict__ wsw,
    _Float16* __restrict__ y1, float* __restrict__ gsum,
    float* __restrict__ gsq) {
  __shared__ __align__(32) _Float16 sA[128 * EDGE_KP];   // 40 KB
  __shared__ __align__(32) _Float16 sB[EDGE_KP * COUT];  // 40 KB, swizzled
  __shared__ int sNbr[128], sCtr[128], sBat[128];
  __shared__ float redS[COUT], redQ[COUT];
  const int tid = threadIdx.x;
  const size_t rowBase = (size_t)blockIdx.x * 128;

  stage_weights(wsw, sB, EDGE_KP * COUT * 2, tid);  // TDM (or vector copy)

  if (tid < 128) {
    const size_t gr = rowBase + tid;
    const int b = (int)(gr / (N_ * KNN));
    const int rem = (int)(gr % (N_ * KNN));
    sBat[tid] = b;
    sCtr[tid] = rem / KNN;
    sNbr[tid] = idx[gr];
  }
  if (tid < COUT) { redS[tid] = 0.f; redQ[tid] = 0.f; }
  __syncthreads();

  // build A tile in 8-element chunks (regions are 8-aligned: 64/128)
  for (int ec = tid; ec < 128 * (EDGE_KP / 8); ec += 256) {
    const int r  = ec / (EDGE_KP / 8);
    const int c0 = (ec % (EDGE_KP / 8)) * 8;
    const int b = sBat[r], n = sCtr[r], nb = sNbr[r];
    v8h v;
    if (c0 < CIN) {
      v = *(const v8h*)(featT + ((size_t)b * N_ + n) * CIN + c0);
    } else if (c0 < 2 * CIN) {
      v = *(const v8h*)(featT + ((size_t)b * N_ + nb) * CIN + (c0 - CIN));
    } else if (c0 == 2 * CIN) {
#pragma unroll
      for (int j = 0; j < 8; ++j) v[j] = (_Float16)0.f;
#pragma unroll
      for (int dd = 0; dd < 3; ++dd)
        v[dd] = (_Float16)(xyz[((size_t)b * 3 + dd) * N_ + nb] -
                           xyz[((size_t)b * 3 + dd) * N_ + n]);
    } else {
#pragma unroll
      for (int j = 0; j < 8; ++j) v[j] = (_Float16)0.f;
    }
    *(v8h*)(sA + r * EDGE_KP + c0) = v;
  }
  __syncthreads();

  const int wv = tid >> 5, lane = tid & 31;
  v8f acc[8];
#pragma unroll
  for (int ct = 0; ct < 8; ++ct)
#pragma unroll
    for (int i = 0; i < 8; ++i) acc[ct][i] = 0.f;

  wmma_block<8, 5, EDGE_KP>(sA, sB, wv, lane, acc);

  const int half = lane >> 4, col0 = lane & 15;
#pragma unroll
  for (int ct = 0; ct < 8; ++ct) {
    float s = 0.f, q = 0.f;
#pragma unroll
    for (int r = 0; r < 8; ++r) {
      const float v = acc[ct][r];
      s += v; q += v * v;
      const size_t gr = rowBase + wv * 16 + half * 8 + r;
      y1[gr * COUT + ct * 16 + col0] = (_Float16)v;
    }
    atomicAdd(&redS[ct * 16 + col0], s);
    atomicAdd(&redQ[ct * 16 + col0], q);
  }
  __syncthreads();
  if (tid < COUT) {
    atomicAdd(&gsum[tid], redS[tid]);
    atomicAdd(&gsq[tid], redQ[tid]);
  }
}

// ---------------------------------------------------------------------------
// GEMM2: relu(bn1(y1)) @ w2^T -> y2 (f16) + stats2 (BN fused into A load).
// ---------------------------------------------------------------------------
__global__ __launch_bounds__(256) void k_gemm2(
    const _Float16* __restrict__ y1, const _Float16* __restrict__ wsw,
    const float* __restrict__ sc1, const float* __restrict__ sh1,
    _Float16* __restrict__ y2, float* __restrict__ gsum,
    float* __restrict__ gsq) {
  __shared__ __align__(32) _Float16 sA[128 * COUT];  // 32 KB
  __shared__ __align__(32) _Float16 sB[COUT * COUT]; // 32 KB, swizzled
  __shared__ float s_sc[COUT], s_sh[COUT], redS[COUT], redQ[COUT];
  const int tid = threadIdx.x;
  const size_t rowBase = (size_t)blockIdx.x * 128;

  stage_weights(wsw, sB, COUT * COUT * 2, tid);

  if (tid < COUT) {
    s_sc[tid] = sc1[tid]; s_sh[tid] = sh1[tid];
    redS[tid] = 0.f;      redQ[tid] = 0.f;
  }
  __syncthreads();

  for (int ec = tid; ec < 128 * (COUT / 8); ec += 256) {
    const int r  = ec >> 4;
    const int c0 = (ec & 15) * 8;
    const v8h yv = *(const v8h*)(y1 + (rowBase + r) * COUT + c0);
    v8h xv;
#pragma unroll
    for (int j = 0; j < 8; ++j)
      xv[j] = (_Float16)fmaxf((float)yv[j] * s_sc[c0 + j] + s_sh[c0 + j], 0.f);
    *(v8h*)(sA + r * COUT + c0) = xv;
  }
  __syncthreads();

  const int wv = tid >> 5, lane = tid & 31;
  v8f acc[8];
#pragma unroll
  for (int ct = 0; ct < 8; ++ct)
#pragma unroll
    for (int i = 0; i < 8; ++i) acc[ct][i] = 0.f;

  wmma_block<8, 4, COUT>(sA, sB, wv, lane, acc);

  const int half = lane >> 4, col0 = lane & 15;
#pragma unroll
  for (int ct = 0; ct < 8; ++ct) {
    float s = 0.f, q = 0.f;
#pragma unroll
    for (int r = 0; r < 8; ++r) {
      const float v = acc[ct][r];
      s += v; q += v * v;
      const size_t gr = rowBase + wv * 16 + half * 8 + r;
      y2[gr * COUT + ct * 16 + col0] = (_Float16)v;
    }
    atomicAdd(&redS[ct * 16 + col0], s);
    atomicAdd(&redQ[ct * 16 + col0], q);
  }
  __syncthreads();
  if (tid < COUT) {
    atomicAdd(&gsum[tid], redS[tid]);
    atomicAdd(&gsq[tid], redQ[tid]);
  }
}

// ---------------------------------------------------------------------------
// GEMM3 (attention hidden): relu(bn2(y2)) @ wa1^T[128x32] -> a1 + stats3.
// ---------------------------------------------------------------------------
__global__ __launch_bounds__(256) void k_gemm3(
    const _Float16* __restrict__ y2, const _Float16* __restrict__ wsw,
    const float* __restrict__ sc2, const float* __restrict__ sh2,
    _Float16* __restrict__ a1, float* __restrict__ gsum,
    float* __restrict__ gsq) {
  __shared__ __align__(32) _Float16 sA[128 * COUT];  // 32 KB
  __shared__ __align__(32) _Float16 sB[COUT * 32];   // 8 KB, swizzled
  __shared__ float s_sc[COUT], s_sh[COUT], redS[32], redQ[32];
  const int tid = threadIdx.x;
  const size_t rowBase = (size_t)blockIdx.x * 128;

  stage_weights(wsw, sB, COUT * 32 * 2, tid);

  if (tid < COUT) { s_sc[tid] = sc2[tid]; s_sh[tid] = sh2[tid]; }
  if (tid < 32)   { redS[tid] = 0.f;      redQ[tid] = 0.f; }
  __syncthreads();

  for (int ec = tid; ec < 128 * (COUT / 8); ec += 256) {
    const int r  = ec >> 4;
    const int c0 = (ec & 15) * 8;
    const v8h yv = *(const v8h*)(y2 + (rowBase + r) * COUT + c0);
    v8h xv;
#pragma unroll
    for (int j = 0; j < 8; ++j)
      xv[j] = (_Float16)fmaxf((float)yv[j] * s_sc[c0 + j] + s_sh[c0 + j], 0.f);
    *(v8h*)(sA + r * COUT + c0) = xv;
  }
  __syncthreads();

  const int wv = tid >> 5, lane = tid & 31;
  v8f acc[2];
#pragma unroll
  for (int ct = 0; ct < 2; ++ct)
#pragma unroll
    for (int i = 0; i < 8; ++i) acc[ct][i] = 0.f;

  wmma_block<2, 4, COUT>(sA, sB, wv, lane, acc);

  const int half = lane >> 4, col0 = lane & 15;
#pragma unroll
  for (int ct = 0; ct < 2; ++ct) {
    float s = 0.f, q = 0.f;
#pragma unroll
    for (int r = 0; r < 8; ++r) {
      const float v = acc[ct][r];
      s += v; q += v * v;
      const size_t gr = rowBase + wv * 16 + half * 8 + r;
      a1[gr * 32 + ct * 16 + col0] = (_Float16)v;
    }
    atomicAdd(&redS[ct * 16 + col0], s);
    atomicAdd(&redQ[ct * 16 + col0], q);
  }
  __syncthreads();
  if (tid < 32) {
    atomicAdd(&gsum[tid], redS[tid]);
    atomicAdd(&gsq[tid], redQ[tid]);
  }
}

// ---------------------------------------------------------------------------
// Fold sum/sumsq into fused BN scale/shift: scale=g*rstd, shift=be-mean*scale
// ---------------------------------------------------------------------------
__global__ void k_finalize(const float* __restrict__ sum,
                           const float* __restrict__ sq,
                           const float* __restrict__ g,
                           const float* __restrict__ be,
                           float* __restrict__ scale,
                           float* __restrict__ shift, int C, float invCount) {
  const int c = threadIdx.x;
  if (c < C) {
    const float m  = sum[c] * invCount;
    const float v  = sq[c] * invCount - m * m;
    const float rs = rsqrtf(v + 1e-5f);
    const float s  = g[c] * rs;
    scale[c] = s;
    shift[c] = be[c] - m * s;
  }
}

// ---------------------------------------------------------------------------
// Final: bn3+relu -> wa2 dot -> softmax over K -> weighted sum of h -> out
// ---------------------------------------------------------------------------
__global__ __launch_bounds__(256) void k_attn_out(
    const _Float16* __restrict__ y2, const _Float16* __restrict__ a1,
    const float* __restrict__ wa2, const float* __restrict__ sc2,
    const float* __restrict__ sh2, const float* __restrict__ sc3,
    const float* __restrict__ sh3, float* __restrict__ out) {
  __shared__ float swa2[32], ssc3[32], ssh3[32], s_s2[COUT], s_h2[COUT];
  const int tid = threadIdx.x;
  if (tid < 32) { swa2[tid] = wa2[tid]; ssc3[tid] = sc3[tid]; ssh3[tid] = sh3[tid]; }
  if (tid < COUT) { s_s2[tid] = sc2[tid]; s_h2[tid] = sh2[tid]; }
  __syncthreads();

  const int t = blockIdx.x * 256 + tid;
  const int b = t / N_, n = t % N_;
  const size_t rbase = (size_t)t * KNN;

  // get the 16 y2 rows (256B each) in flight while scores/softmax compute
#pragma unroll
  for (int kk = 0; kk < KNN; ++kk) {
    __builtin_prefetch(y2 + (rbase + kk) * COUT, 0, 3);
    __builtin_prefetch(y2 + (rbase + kk) * COUT + 64, 0, 3);
  }

  float sc[KNN];
#pragma unroll
  for (int kk = 0; kk < KNN; ++kk) {
    float s = 0.f;
    const _Float16* ar = a1 + (rbase + kk) * 32;
#pragma unroll
    for (int q = 0; q < 4; ++q) {
      const v8h av = *(const v8h*)(ar + q * 8);
#pragma unroll
      for (int j = 0; j < 8; ++j) {
        const int c = q * 8 + j;
        s += swa2[c] * fmaxf((float)av[j] * ssc3[c] + ssh3[c], 0.f);
      }
    }
    sc[kk] = s;
  }
  float mx = sc[0];
#pragma unroll
  for (int kk = 1; kk < KNN; ++kk) mx = fmaxf(mx, sc[kk]);
  float den = 0.f;
#pragma unroll
  for (int kk = 0; kk < KNN; ++kk) { sc[kk] = __expf(sc[kk] - mx); den += sc[kk]; }
  const float inv = 1.f / den;

  for (int c8 = 0; c8 < COUT / 8; ++c8) {
    float accv[8];
#pragma unroll
    for (int j = 0; j < 8; ++j) accv[j] = 0.f;
#pragma unroll
    for (int kk = 0; kk < KNN; ++kk) {
      const v8h hv = *(const v8h*)(y2 + (rbase + kk) * COUT + c8 * 8);
#pragma unroll
      for (int j = 0; j < 8; ++j)
        accv[j] += sc[kk] *
                   fmaxf((float)hv[j] * s_s2[c8 * 8 + j] + s_h2[c8 * 8 + j], 0.f);
    }
#pragma unroll
    for (int j = 0; j < 8; ++j)
      out[((size_t)b * COUT + c8 * 8 + j) * N_ + n] = accv[j] * inv;
  }
}

// ---------------------------------------------------------------------------
extern "C" void kernel_launch(void* const* d_in, const int* in_sizes, int n_in,
                              void* d_out, int out_size, void* d_ws,
                              size_t ws_size, hipStream_t stream) {
  (void)in_sizes; (void)n_in; (void)out_size; (void)ws_size;
  const float* xyz  = (const float*)d_in[0];
  const float* feat = (const float*)d_in[1];
  const float* w1   = (const float*)d_in[2];
  const float* g1   = (const float*)d_in[4];
  const float* be1  = (const float*)d_in[5];
  const float* w2   = (const float*)d_in[6];
  const float* g2   = (const float*)d_in[8];
  const float* be2  = (const float*)d_in[9];
  const float* wa1  = (const float*)d_in[10];
  const float* ga1  = (const float*)d_in[12];
  const float* bea1 = (const float*)d_in[13];
  const float* wa2  = (const float*)d_in[14];
  float* out = (float*)d_out;
  char* ws = (char*)d_ws;

  // workspace layout
  int*      idx   = (int*)(ws + 0);                      // 2 MB
  _Float16* featT = (_Float16*)(ws + ((size_t)2 << 20)); // 4 MB
  float*    stats = (float*)(ws + ((size_t)6 << 20));    // 1216 floats
  float* sum1 = stats;        float* sq1 = stats + 128;
  float* sum2 = stats + 256;  float* sq2 = stats + 384;
  float* sum3 = stats + 512;  float* sq3 = stats + 544;
  float* sc1 = stats + 576;   float* sh1 = stats + 704;
  float* sc2 = stats + 832;   float* sh2 = stats + 960;
  float* sc3 = stats + 1088;  float* sh3 = stats + 1120;
  _Float16* wsw1 = (_Float16*)(ws + ((size_t)7 << 20));                 // 40 KB
  _Float16* wsw2 = (_Float16*)(ws + ((size_t)7 << 20) + (64u << 10));   // 32 KB
  _Float16* wsw3 = (_Float16*)(ws + ((size_t)7 << 20) + (128u << 10));  // 8 KB
  _Float16* y1 = (_Float16*)(ws + ((size_t)8 << 20));                   // 128 MB
  _Float16* y2 = (_Float16*)(ws + ((size_t)8 << 20) + (size_t)MROWS * COUT * 2);
  _Float16* a1 = y1;  // y1 is dead after gemm2 -> reuse for attention hidden

  const float invM = 1.0f / (float)MROWS;

  k_zero<<<8, 256, 0, stream>>>(stats, 1216);
  k_prepw<<<(EDGE_KP * COUT + 255) / 256, 256, 0, stream>>>(
      w1, wsw1, 8, EDGE_K, EDGE_KP * COUT);
  k_prepw<<<(COUT * COUT + 255) / 256, 256, 0, stream>>>(
      w2, wsw2, 8, COUT, COUT * COUT);
  k_prepw<<<(COUT * 32 + 255) / 256, 256, 0, stream>>>(
      wa1, wsw3, 2, COUT, COUT * 32);
  k_transpose<<<(B_ * N_ * CIN + 255) / 256, 256, 0, stream>>>(feat, featT);
  k_knn<<<B_ * (N_ / 256), 256, 0, stream>>>(xyz, idx);

  k_gemm1<<<MROWS / 128, 256, 0, stream>>>(featT, xyz, idx, wsw1, y1, sum1, sq1);
  k_finalize<<<1, 128, 0, stream>>>(sum1, sq1, g1, be1, sc1, sh1, 128, invM);

  k_gemm2<<<MROWS / 128, 256, 0, stream>>>(y1, wsw2, sc1, sh1, y2, sum2, sq2);
  k_finalize<<<1, 128, 0, stream>>>(sum2, sq2, g2, be2, sc2, sh2, 128, invM);

  k_gemm3<<<MROWS / 128, 256, 0, stream>>>(y2, wsw3, sc2, sh2, a1, sum3, sq3);
  k_finalize<<<1, 128, 0, stream>>>(sum3, sq3, ga1, bea1, sc3, sh3, 32, invM);

  k_attn_out<<<B_ * N_ / 256, 256, 0, stream>>>(y2, a1, wa2, sc2, sh2, sc3,
                                                sh3, out);
}
